// TemporalFusion_36601711296538
// MI455X (gfx1250) — matrix operand order repaired and checked by reading, not compiled, not run on hardware
//
#include <hip/hip_runtime.h>
#include <hip/hip_bf16.h>

typedef __attribute__((ext_vector_type(16))) _Float16 v16h;
typedef __attribute__((ext_vector_type(8)))  _Float16 v8h;
typedef __attribute__((ext_vector_type(8)))  float    v8f;
typedef unsigned int u32x4 __attribute__((ext_vector_type(4)));
typedef int          i32x4 __attribute__((ext_vector_type(4)));
typedef int          i32x8 __attribute__((ext_vector_type(8)));

#define HDIM 64
#define WDIM 64
#define HW   4096
#define HP   66            // haloed spatial extent
#define CCH  256
#define BATCH 4
#define NGROUPS 4
#define CORR_CH 49
#define CIN_OFF 305
#define CIN_OFF_PAD 320
#define NOFF 72
#define NOFF_PAD 80
#define CIN_FUS 1024
#define FUS_CHUNKS (9 * (CIN_FUS / 32))   // 288

#ifndef __has_builtin
#define __has_builtin(x) 0
#endif
#if __has_builtin(__builtin_amdgcn_tensor_load_to_lds)
#define USE_TDM 1
#else
#define USE_TDM 0
#endif

// ---------------------------------------------------------------------------
// WMMA wrapper: D = A(16x32 f16) * B(32x16 f16) + C(16x16 f32)
// ---------------------------------------------------------------------------
__device__ __forceinline__ v8f wmma_f16(v16h a, v16h b, v8f c) {
  return __builtin_amdgcn_wmma_f32_16x16x32_f16(
      false, a, false, b, (short)0, c, false, false);
}

// ---------------------------------------------------------------------------
// A-fragment loader over HALOED NHWC f16 input [frame][66][66][cinpad]:
// output pixel (h, w) + tap (ky,kx) -> haloed coords (h+ky, w+kx), always
// in-bounds, so the load is unconditional (no exec-mask dance, no re-zeroing).
// 16-bit A 16x32 layout (ISA 7.12.2): lane<16 holds M=lane, K {0..7,16..23};
// lane>=16 holds M=lane-16, K {8..15,24..31}: two contiguous 16B loads.
// ---------------------------------------------------------------------------
__device__ __forceinline__ v16h load_a_halo(const _Float16* __restrict__ xin,
                                            int frame, int h, int w0, int lane,
                                            int cinpad, int c0, int ky, int kx) {
  const int m = lane & 15;
  const _Float16* p = xin +
      (((size_t)frame * HP + (h + ky)) * HP + (w0 + m + kx)) * cinpad +
      c0 + ((lane & 16) ? 8 : 0);
  const v8h lo = *(const v8h*)p;
  const v8h hi = *(const v8h*)(p + 16);
  v16h a;
#pragma unroll
  for (int i = 0; i < 8; ++i) { a[i] = lo[i]; a[8 + i] = hi[i]; }
  return a;
}

// ---------------------------------------------------------------------------
// TDM: stage one 32-row x 256-col f16 tile (row stride 256) into LDS.
// D# per CDNA5 ISA: group0 = {count=1, lds_addr,
// global_addr[56:0], type=2}; group1 = {data_size=2B, tensor_dim0=256,
// tensor_dim1=9216, tile_dim0=256, tile_dim1=32, tensor_dim0_stride=256}.
// ---------------------------------------------------------------------------
#if USE_TDM
__device__ __forceinline__ void tdm_stage_b(const _Float16* gsrc, unsigned lds_bytes) {
  const unsigned long long ga = (unsigned long long)(size_t)gsrc;
  u32x4 g0;
  g0[0] = 1u;                                        // count=1, user descriptor
  g0[1] = lds_bytes;                                 // lds_addr (bytes)
  g0[2] = (unsigned)(ga & 0xffffffffu);              // global_addr[31:0]
  g0[3] = (unsigned)((ga >> 32) & 0x1ffffffu)        // global_addr[56:32]
        | (2u << 30);                                // type=2 (image)
  i32x8 g1;
  g1[0] = 0x00010000;                                // wg_mask=0, data_size=1 (2B)
  g1[1] = (int)(256u << 16);                         // tensor_dim0[15:0]=256
  g1[2] = (int)(9216u << 16);                        // tensor_dim0[31:16]=0, tensor_dim1[15:0]=9216
  g1[3] = (int)(256u << 16);                         // tensor_dim1[31:16]=0, tile_dim0=256
  g1[4] = 32;                                        // tile_dim1=32, tile_dim2=0
  g1[5] = 256;                                       // tensor_dim0_stride[31:0]
  g1[6] = 0;                                         // stride hi, dim1_stride lo
  g1[7] = 0;
  i32x4 z4 = {0, 0, 0, 0};
#if __clang_major__ >= 23
  i32x8 z8 = {0, 0, 0, 0, 0, 0, 0, 0};
  __builtin_amdgcn_tensor_load_to_lds(g0, g1, z4, z4, z8, 0);
#else
  __builtin_amdgcn_tensor_load_to_lds(g0, g1, z4, z4, 0);
#endif
}
#endif

// ---------------------------------------------------------------------------
// Prep kernels
// ---------------------------------------------------------------------------
__global__ void zero_f16_kernel(_Float16* __restrict__ p, long long n) {
  long long i = (long long)blockIdx.x * blockDim.x + threadIdx.x;
  if (i < n) p[i] = (_Float16)0.0f;
}

__global__ void pack_x_kernel(const float* __restrict__ in, _Float16* __restrict__ xo) {
  size_t idx = (size_t)blockIdx.x * blockDim.x + threadIdx.x;
  if (idx >= (size_t)16 * HW * CCH) return;
  int c = (int)(idx % CCH);
  size_t r = idx / CCH;
  int pix = (int)(r % HW);
  int f = (int)(r / HW);
  xo[idx] = (_Float16)in[((size_t)f * CCH + c) * HW + pix];
}

__global__ void pack_woff_kernel(const float* __restrict__ w, _Float16* __restrict__ wo) {
  size_t idx = (size_t)blockIdx.x * blockDim.x + threadIdx.x;
  if (idx >= (size_t)9 * CIN_OFF_PAD * NOFF_PAD) return;
  int o  = (int)(idx % NOFF_PAD);
  int ci = (int)((idx / NOFF_PAD) % CIN_OFF_PAD);
  int t  = (int)(idx / ((size_t)NOFF_PAD * CIN_OFF_PAD));
  float v = 0.0f;
  if (o < NOFF && ci < CIN_OFF) v = w[((size_t)o * CIN_OFF + ci) * 9 + t];
  wo[idx] = (_Float16)v;
}

__global__ void pack_wdef_kernel(const float* __restrict__ w, _Float16* __restrict__ wo) {
  // dst: [g][tap][cg][o] = w_def[o][g*64+cg][tap]
  size_t idx = (size_t)blockIdx.x * blockDim.x + threadIdx.x;
  if (idx >= (size_t)NGROUPS * 9 * 64 * CCH) return;
  int o  = (int)(idx % CCH);
  int cg = (int)((idx / CCH) % 64);
  int t  = (int)((idx / ((size_t)CCH * 64)) % 9);
  int g  = (int)(idx / ((size_t)CCH * 64 * 9));
  wo[idx] = (_Float16)w[((size_t)o * CCH + g * 64 + cg) * 9 + t];
}

__global__ void pack_wfus_kernel(const float* __restrict__ w, _Float16* __restrict__ wo) {
  // dst: [tap][ci][o] = w_fusion[o][ci][tap]
  size_t idx = (size_t)blockIdx.x * blockDim.x + threadIdx.x;
  if (idx >= (size_t)9 * CIN_FUS * CCH) return;
  int o  = (int)(idx % CCH);
  int ci = (int)((idx / CCH) % CIN_FUS);
  int t  = (int)(idx / ((size_t)CCH * CIN_FUS));
  wo[idx] = (_Float16)w[((size_t)o * CIN_FUS + ci) * 9 + t];
}

// ---------------------------------------------------------------------------
// Stage 1a: x channels -> haloed offset-conv input [b][66][66][320]
// ---------------------------------------------------------------------------
__global__ void concat_x_kernel(const _Float16* __restrict__ x16,
                                _Float16* __restrict__ xc, int ii) {
  size_t idx = (size_t)blockIdx.x * blockDim.x + threadIdx.x;
  if (idx >= (size_t)BATCH * HW * CIN_OFF_PAD) return;
  int c = (int)(idx % CIN_OFF_PAD);
  size_t r = idx / CIN_OFF_PAD;
  int pix = (int)(r % HW);
  int b = (int)(r / HW);
  int h = pix >> 6, w = pix & 63;
  _Float16* dst = xc + (((size_t)b * HP + h + 1) * HP + w + 1) * CIN_OFF_PAD;
  if (c < CCH) {
    dst[c] = x16[((size_t)(b * NGROUPS + ii) * HW + pix) * CCH + c];
  } else if (c >= CIN_OFF) {
    dst[c] = (_Float16)0.0f;
  }
}

// ---------------------------------------------------------------------------
// Stage 1b: correlation -> channels 256..304 of haloed offset-conv input
// ---------------------------------------------------------------------------
__global__ void corr_kernel(const _Float16* __restrict__ x16,
                            _Float16* __restrict__ xc, int ii) {
  int idx = blockIdx.x * blockDim.x + threadIdx.x;
  if (idx >= BATCH * HW * CORR_CH) return;
  int d   = idx % CORR_CH;
  int pix = (idx / CORR_CH) % HW;
  int b   = idx / (CORR_CH * HW);
  int di = d / 7 - 3, dj = d % 7 - 3;
  int h = pix >> 6, w = pix & 63;
  int hs = h + 2 * di, ws = w + 2 * dj;
  float s = 0.0f;
  if (hs >= 0 && hs < HDIM && ws >= 0 && ws < WDIM) {
    const _Float16* xa = x16 + ((size_t)(b * NGROUPS + ii) * HW + pix) * CCH;
    const _Float16* xb = x16 + ((size_t)(b * NGROUPS + 3) * HW + hs * WDIM + ws) * CCH;
#pragma unroll 4
    for (int c = 0; c < CCH; c += 8) {
      v8h va = *(const v8h*)(xa + c);
      v8h vb = *(const v8h*)(xb + c);
#pragma unroll
      for (int j = 0; j < 8; ++j) s += (float)va[j] * (float)vb[j];
    }
  }
  xc[(((size_t)b * HP + h + 1) * HP + w + 1) * CIN_OFF_PAD + CCH + d] =
      (_Float16)(s * (1.0f / 256.0f));
}

// ---------------------------------------------------------------------------
// Stage 2: offset conv (Cin 320 padded, Cout 72) via WMMA implicit GEMM.
// ---------------------------------------------------------------------------
__global__ void __launch_bounds__(128) offset_conv_kernel(
    const _Float16* __restrict__ xc, const _Float16* __restrict__ wp,
    float* __restrict__ off) {
  const int lane  = threadIdx.x & 31;
  const int gwave = blockIdx.x * 4 + (threadIdx.x >> 5);   // 0..5119
  const int ng = gwave % 5;
  int t = gwave / 5;                                        // 0..1023
  const int wt = t & 3;  t >>= 2;
  const int h  = t & 63; t >>= 6;
  const int b  = t;
  const int w0 = wt * 16, n0 = ng * 16;
  v8f acc = {};
  for (int tap = 0; tap < 9; ++tap) {
    const int ky = tap / 3, kx = tap % 3;
    const _Float16* wtb = wp + (size_t)tap * CIN_OFF_PAD * NOFF_PAD;
    for (int c0 = 0; c0 < CIN_OFF_PAD; c0 += 32) {
      v16h a  = load_a_halo(xc, b, h, w0, lane, CIN_OFF_PAD, c0, ky, kx);
      v16h bf;
#pragma unroll
      for (int s = 0; s < 16; ++s)
        bf[s] = wtb[(size_t)(c0 + ((lane & 16) ? 8 : 0) + s + (s >= 8 ? 8 : 0)) * NOFF_PAD + n0 + (lane & 15)];
      acc = wmma_f16(a, bf, acc);
    }
  }
  const int col = n0 + (lane & 15);
  if (col < NOFF) {
    const int mb = (lane & 16) ? 8 : 0;
#pragma unroll
    for (int i = 0; i < 8; ++i) {
      const int pix = h * WDIM + w0 + mb + i;
      off[((size_t)b * HW + pix) * NOFF + col] = acc[i];
    }
  }
}

// ---------------------------------------------------------------------------
// Stage 3: deformable conv v1, fused bilinear im2col + WMMA GEMM.
// One wave: 16 pixels x 64 out channels (4 accumulators amortize the costly
// sampled A-fragment). Writes f16 into haloed fusion concat buffer.
// ---------------------------------------------------------------------------
__global__ void __launch_bounds__(128) deform_conv_kernel(
    const _Float16* __restrict__ x16, const float* __restrict__ off,
    const _Float16* __restrict__ wp, _Float16* __restrict__ xfus, int ii) {
  const int lane  = threadIdx.x & 31;
  const int gwave = blockIdx.x * 4 + (threadIdx.x >> 5);   // 0..4095
  const int ng = gwave & 3; int t = gwave >> 2;
  const int wt = t & 3;  t >>= 2;
  const int h  = t & 63; t >>= 6;
  const int b  = t;
  const int w0 = wt * 16, n0 = ng * 64;
  const int m  = lane & 15;
  const int w  = w0 + m;
  const int pix = h * WDIM + w;
  const int hb = (lane & 16) ? 8 : 0;
  const _Float16* xb = x16 + (size_t)(b * NGROUPS + ii) * HW * CCH;
  const float*    ob = off + ((size_t)b * HW + pix) * NOFF;
  v8f acc0 = {}, acc1 = {}, acc2 = {}, acc3 = {};
  for (int g = 0; g < NGROUPS; ++g) {
    for (int tap = 0; tap < 9; ++tap) {
      const float oy  = ob[(g * 9 + tap) * 2 + 0];
      const float ox  = ob[(g * 9 + tap) * 2 + 1];
      const float py  = (float)(h + tap / 3 - 1) + oy;
      const float px  = (float)(w + tap % 3 - 1) + ox;
      const float y0f = floorf(py), x0f = floorf(px);
      const float fy = py - y0f, fx = px - x0f;
      const int y0 = (int)y0f, x0 = (int)x0f;
      for (int c0 = 0; c0 < 64; c0 += 32) {
        const int cb = g * 64 + c0 + hb;
        float af[16];
#pragma unroll
        for (int i = 0; i < 16; ++i) af[i] = 0.0f;
#pragma unroll
        for (int dy = 0; dy < 2; ++dy) {
#pragma unroll
          for (int dx = 0; dx < 2; ++dx) {
            const int yy = y0 + dy, xx = x0 + dx;
            if (yy < 0 || yy >= HDIM || xx < 0 || xx >= WDIM) continue;
            const float wg = (dy ? fy : 1.0f - fy) * (dx ? fx : 1.0f - fx);
            const _Float16* p = xb + ((size_t)yy * WDIM + xx) * CCH + cb;
            const v8h v0 = *(const v8h*)p;
            const v8h v1 = *(const v8h*)(p + 16);
#pragma unroll
            for (int i = 0; i < 8; ++i) {
              af[i]     += wg * (float)v0[i];
              af[8 + i] += wg * (float)v1[i];
            }
          }
        }
        v16h a;
#pragma unroll
        for (int i = 0; i < 16; ++i) a[i] = (_Float16)af[i];
        const _Float16* wrow = wp + ((size_t)((g * 9 + tap) * 64 + c0)) * CCH + n0 + m;
        __builtin_prefetch((const void*)(wrow + 32 * CCH), 0, 0);
        v16h b0, b1, b2, b3;
#pragma unroll
        for (int s = 0; s < 16; ++s) {
          const _Float16* q = wrow + (size_t)(hb + s + (s >= 8 ? 8 : 0)) * CCH;
          b0[s] = q[0]; b1[s] = q[16]; b2[s] = q[32]; b3[s] = q[48];
        }
        acc0 = wmma_f16(a, b0, acc0);
        acc1 = wmma_f16(a, b1, acc1);
        acc2 = wmma_f16(a, b2, acc2);
        acc3 = wmma_f16(a, b3, acc3);
      }
    }
  }
  const int col = lane & 15;
#pragma unroll
  for (int i = 0; i < 8; ++i) {
    _Float16* dst = xfus + (((size_t)b * HP + h + 1) * HP + (w0 + hb + i + 1)) * CIN_FUS
                  + ii * CCH + n0;
    dst[col]      = (_Float16)acc0[i];
    dst[16 + col] = (_Float16)acc1[i];
    dst[32 + col] = (_Float16)acc2[i];
    dst[48 + col] = (_Float16)acc3[i];
  }
}

// ---------------------------------------------------------------------------
// Copy reference frame y into haloed fusion concat channels 768..1023
// ---------------------------------------------------------------------------
__global__ void copy_y_kernel(const _Float16* __restrict__ x16,
                              _Float16* __restrict__ xfus) {
  size_t idx = (size_t)blockIdx.x * blockDim.x + threadIdx.x;
  if (idx >= (size_t)BATCH * HW * CCH) return;
  int c = (int)(idx % CCH);
  size_t r = idx / CCH;
  int pix = (int)(r % HW);
  int b = (int)(r / HW);
  int h = pix >> 6, w = pix & 63;
  xfus[(((size_t)b * HP + h + 1) * HP + w + 1) * CIN_FUS + 3 * CCH + c] =
      x16[((size_t)(b * NGROUPS + 3) * HW + pix) * CCH + c];
}

// ---------------------------------------------------------------------------
// Stage 4: fusion conv (Cin 1024, Cout 256) via WMMA implicit GEMM.
// Block = 4 waves sharing one 16-pixel tile, covering all 256 out channels.
// The 32x256 f16 B-tile per K-chunk is staged into LDS once per block —
// via the Tensor Data Mover (double buffered, s_wait_tensorcnt) when
// available, else a cooperative vector copy. Waves read B from LDS as single f16 elements.
// ---------------------------------------------------------------------------
__global__ void __launch_bounds__(128) fusion_conv_kernel(
    const _Float16* __restrict__ xin, const _Float16* __restrict__ wp,
    float* __restrict__ out) {
  __shared__ __align__(128) _Float16 bsm[2][32][256];   // 32 KB, double buffered
  const int lane = threadIdx.x & 31;
  const int wv   = threadIdx.x >> 5;    // 0..3
  int t = blockIdx.x;                   // 0..1023 pixel tiles
  const int wt = t & 3;  t >>= 2;
  const int h  = t & 63; t >>= 6;
  const int b  = t;
  const int w0 = wt * 16;
  const int n0 = wv * 64;
  v8f acc0 = {}, acc1 = {}, acc2 = {}, acc3 = {};

#if USE_TDM
  const unsigned lds0 = (unsigned)(size_t)&bsm[0][0][0];
  const unsigned lds1 = (unsigned)(size_t)&bsm[1][0][0];
  if (threadIdx.x < 32) tdm_stage_b(wp, lds0);          // prologue: chunk 0
#endif

  for (int idx = 0; idx < FUS_CHUNKS; ++idx) {
    const int buf = idx & 1;
#if USE_TDM
    if (threadIdx.x < 32) {
      if (idx + 1 < FUS_CHUNKS) {
        tdm_stage_b(wp + (size_t)(idx + 1) * 32 * 256, buf ? lds0 : lds1);
        __builtin_amdgcn_s_wait_tensorcnt(1);           // chunk idx landed
      } else {
        __builtin_amdgcn_s_wait_tensorcnt(0);
      }
    }
    __syncthreads();
#else
    {   // cooperative stage of chunk idx: 512 x v16h, 4 per thread
      const v16h* s = (const v16h*)(wp + (size_t)idx * 32 * 256);
      v16h* d = (v16h*)&bsm[buf][0][0];
#pragma unroll
      for (int j = 0; j < 4; ++j) d[threadIdx.x + j * 128] = s[threadIdx.x + j * 128];
    }
    __syncthreads();
#endif
    const int tap = idx >> 5;
    const int c0  = (idx & 31) * 32;
    v16h a = load_a_halo(xin, b, h, w0, lane, CIN_FUS, c0, tap / 3, tap % 3);
    const _Float16* brow = &bsm[buf][0][n0 + (lane & 15)];
    v16h b0, b1, b2, b3;
#pragma unroll
    for (int s = 0; s < 16; ++s) {
      const _Float16* q = brow + (((lane & 16) ? 8 : 0) + s + (s >= 8 ? 8 : 0)) * 256;
      b0[s] = q[0]; b1[s] = q[16]; b2[s] = q[32]; b3[s] = q[48];
    }
    acc0 = wmma_f16(a, b0, acc0);
    acc1 = wmma_f16(a, b1, acc1);
    acc2 = wmma_f16(a, b2, acc2);
    acc3 = wmma_f16(a, b3, acc3);
    __syncthreads();   // all waves done with bsm[buf] before it is re-staged
  }

  const int col = lane & 15;
  const int mb = (lane & 16) ? 8 : 0;
#pragma unroll
  for (int i = 0; i < 8; ++i) {
    const int pix = h * WDIM + w0 + mb + i;
    out[((size_t)b * CCH + n0 + col) * HW + pix]      = acc0[i];
    out[((size_t)b * CCH + n0 + 16 + col) * HW + pix] = acc1[i];
    out[((size_t)b * CCH + n0 + 32 + col) * HW + pix] = acc2[i];
    out[((size_t)b * CCH + n0 + 48 + col) * HW + pix] = acc3[i];
  }
}

// ---------------------------------------------------------------------------
// Host launcher
// ---------------------------------------------------------------------------
extern "C" void kernel_launch(void* const* d_in, const int* in_sizes, int n_in,
                              void* d_out, int out_size, void* d_ws, size_t ws_size,
                              hipStream_t stream) {
  (void)in_sizes; (void)n_in; (void)out_size; (void)ws_size;
  const float* inputs = (const float*)d_in[0];
  const float* w_off  = (const float*)d_in[1];
  const float* w_def  = (const float*)d_in[2];
  const float* w_fus  = (const float*)d_in[3];
  float* out = (float*)d_out;

  char* ws = (char*)d_ws;
  size_t o = 0;
  _Float16* x16   = (_Float16*)(ws + o); o += (size_t)16 * HW * CCH * 2;               // 32 MB
  _Float16* xfus  = (_Float16*)(ws + o); o += (size_t)BATCH * HP * HP * CIN_FUS * 2;   // 35.7 MB haloed
  _Float16* xc16  = (_Float16*)(ws + o); o += (size_t)BATCH * HP * HP * CIN_OFF_PAD * 2; // 11.2 MB haloed
  float*    offb  = (float*)   (ws + o); o += (size_t)BATCH * HW * NOFF * 4;           // 4.7 MB
  _Float16* woffp = (_Float16*)(ws + o); o += (size_t)9 * CIN_OFF_PAD * NOFF_PAD * 2;
  _Float16* wdefp = (_Float16*)(ws + o); o += (size_t)NGROUPS * 9 * 64 * CCH * 2;
  _Float16* wfusp = (_Float16*)(ws + o); o += (size_t)9 * CIN_FUS * CCH * 2;

  const long long n_xfus = (long long)BATCH * HP * HP * CIN_FUS;
  const long long n_xc   = (long long)BATCH * HP * HP * CIN_OFF_PAD;

  // Zero haloed buffers (halo must be exact zeros every call), then prep.
  zero_f16_kernel <<<(unsigned)((n_xfus + 255) / 256), 256, 0, stream>>>(xfus, n_xfus);
  zero_f16_kernel <<<(unsigned)((n_xc   + 255) / 256), 256, 0, stream>>>(xc16, n_xc);
  pack_x_kernel   <<<65536, 256, 0, stream>>>(inputs, x16);
  pack_woff_kernel<<<  900, 256, 0, stream>>>(w_off, woffp);
  pack_wdef_kernel<<< 2304, 256, 0, stream>>>(w_def, wdefp);
  pack_wfus_kernel<<< 9216, 256, 0, stream>>>(w_fus, wfusp);
  copy_y_kernel   <<<16384, 256, 0, stream>>>(x16, xfus);

  for (int ii = 0; ii < 3; ++ii) {
    concat_x_kernel   <<<20480, 256, 0, stream>>>(x16, xc16, ii);
    corr_kernel       <<< 3136, 256, 0, stream>>>(x16, xc16, ii);
    offset_conv_kernel<<< 1280, 128, 0, stream>>>(xc16, woffp, offb);
    deform_conv_kernel<<< 1024, 128, 0, stream>>>(x16, offb, wdefp, xfus, ii);
  }
  fusion_conv_kernel  <<< 1024, 128, 0, stream>>>(xfus, wfusp, out);
}
